// ARLanguageModel_79577154060448
// MI455X (gfx1250) — matrix operand order repaired and checked by reading, not compiled, not run on hardware
//
#include <hip/hip_runtime.h>
#include <hip/hip_bf16.h>
#include <math.h>

// ---------------- model constants ----------------
#define VD   1024      // D (n_embd)
#define VFF  4096      // FF
#define VNH  16        // heads
#define VDH  64        // head dim
#define VB   2
#define VL   1024
#define VNL  8
#define VVOC 32000
#define VM   (VB*VL)   // 2048 token rows

typedef __bf16 bht;
typedef __attribute__((ext_vector_type(16))) __bf16 v16bf;
typedef __attribute__((ext_vector_type(8)))  float  v8f;
typedef __attribute__((ext_vector_type(4)))  unsigned u32x4;
typedef __attribute__((ext_vector_type(8)))  int      i32x8;
typedef __attribute__((ext_vector_type(4)))  int      i32x4;

union AFrag { v16bf v; unsigned u[8]; };
union BFrag { v16bf v; unsigned u[8]; uint4 q[2]; };

#define WMMA_BF16(A_, B_, C_) \
  __builtin_amdgcn_wmma_f32_16x16x32_bf16(false, (A_), false, (B_), (short)0, (C_), false, false)

// ---------------- embedding + sinusoidal PE ----------------
__global__ __launch_bounds__(256)
void gpt_embed(const int* __restrict__ idx, const float* __restrict__ tok,
               float* __restrict__ x)
{
    const int row = blockIdx.x;            // b*L + l
    const int tid = threadIdx.x;
    const int l   = row & (VL - 1);
    const int t   = idx[row];
    #pragma unroll
    for (int j = 0; j < 4; ++j) {
        int d  = tid + j * 256;
        int i2 = d & ~1;
        float freq = __expf(-(float)i2 * (9.210340371976184f / (float)VD));
        float ang  = (float)l * freq;
        float pe   = (d & 1) ? __cosf(ang) : __sinf(ang);
        x[(size_t)row * VD + d] = tok[(size_t)t * VD + d] + pe;
    }
}

// ---------------- LayerNorm: fp32 in -> bf16 out ----------------
__global__ __launch_bounds__(256)
void gpt_ln(const float* __restrict__ x, const float* __restrict__ w,
            const float* __restrict__ b, bht* __restrict__ out)
{
    __shared__ float sm[8], sq[8];
    const int row = blockIdx.x, tid = threadIdx.x;
    float vals[4], s = 0.f, s2 = 0.f;
    #pragma unroll
    for (int j = 0; j < 4; ++j) {
        float v = x[(size_t)row * VD + tid + j * 256];
        vals[j] = v; s += v; s2 += v * v;
    }
    #pragma unroll
    for (int o = 16; o > 0; o >>= 1) { s += __shfl_xor(s, o, 32); s2 += __shfl_xor(s2, o, 32); }
    if ((tid & 31) == 0) { sm[tid >> 5] = s; sq[tid >> 5] = s2; }
    __syncthreads();
    s = 0.f; s2 = 0.f;
    #pragma unroll
    for (int i = 0; i < 8; ++i) { s += sm[i]; s2 += sq[i]; }
    float mean = s * (1.0f / VD);
    float var  = s2 * (1.0f / VD) - mean * mean;
    float rst  = rsqrtf(var + 1e-5f);
    #pragma unroll
    for (int j = 0; j < 4; ++j) {
        int col = tid + j * 256;
        out[(size_t)row * VD + col] = (bht)((vals[j] - mean) * rst * w[col] + b[col]);
    }
}

// ---------------- WMMA GEMM: C = A(bf16)[M,K] @ W(f32->bf16)[K,N] ----------------
// block = 128 threads (4 waves), tile 128x64, k-step 32, double-buffered pipeline:
//   A tile: Tensor Data Mover with LDS padding (row stride 40 bf16), wave 0 issues.
//   W tile: fp32 global loads issued before the WMMA block, cvt+ds_store after it.
enum { M_BF16 = 0, M_F32 = 1, M_RES = 2, M_RES_BIAS = 3, M_GELU = 4 };

template<int MODE>
__global__ __launch_bounds__(128)
void gpt_gemm(const bht* __restrict__ A, int K,
              const float* __restrict__ W, int N,
              const float* __restrict__ bias, const float* __restrict__ res,
              float* __restrict__ outF, bht* __restrict__ outB)
{
    __shared__ bht Al[2][128 * 40];  // [m][k], stride 40 bf16 (TDM pad: +4dw / 16dw)
    __shared__ bht Bl[2][64 * 40];   // [n][k] = W^T tile

    const int tid  = threadIdx.x;
    const int wave = tid >> 5, lane = tid & 31, half = lane >> 4, lm = lane & 15;
    const int n0 = blockIdx.x * 64, m0 = blockIdx.y * 128;

    // ---- TDM descriptor group1 (uniform, constant across k-steps) ----
    const unsigned ldsA[2] = { (unsigned)(uintptr_t)(void*)&Al[0][0],
                               (unsigned)(uintptr_t)(void*)&Al[1][0] };
    i32x8 g1;
    g1[0] = 0x06D10000;                        // data_size=2B, pad_enable, 16dw interval, 4dw pad
    g1[1] = (K & 0xFFFF) << 16;                // tensor_dim0[15:0]
    g1[2] = ((unsigned)K >> 16) | (VM << 16);  // tensor_dim0[31:16] | tensor_dim1[15:0]
    g1[3] = 32 << 16;                          // tensor_dim1[31:16]=0 | tile_dim0=32
    g1[4] = 128;                               // tile_dim1=128, tile_dim2=0
    g1[5] = K;                                 // tensor_dim0_stride (elements)
    g1[6] = 0;
    g1[7] = 0;
    const i32x4 gz4 = {0, 0, 0, 0};
    const i32x8 gz8 = {0, 0, 0, 0, 0, 0, 0, 0};

    float4 wr[4];  // in-flight W registers

    auto issueA = [&](int kk, int buf) {
        if (tid < 32) {   // wave-uniform branch, wave 0 only
            unsigned long long ga =
                (unsigned long long)(const void*)(A + (size_t)m0 * K + kk);
            u32x4 g0;
            g0[0] = 1u;                                 // count=1
            g0[1] = ldsA[buf];                          // lds_addr
            g0[2] = (unsigned)ga;                       // global_addr lo
            g0[3] = (unsigned)(ga >> 32) | (2u << 30);  // global_addr hi | type=2
            __builtin_amdgcn_tensor_load_to_lds(g0, g1, gz4, gz4, gz8, 0);
        }
    };
    auto loadW = [&](int kk) {
        #pragma unroll
        for (int i = 0; i < 4; ++i) {
            int c = i * 128 + tid;
            int k = c >> 4, n4 = (c & 15) * 4;
            wr[i] = *(const float4*)(W + (size_t)(kk + k) * N + n0 + n4);
        }
    };
    auto storeW = [&](int buf) {
        #pragma unroll
        for (int i = 0; i < 4; ++i) {
            int c = i * 128 + tid;
            int k = c >> 4, n4 = (c & 15) * 4;
            Bl[buf][(n4 + 0) * 40 + k] = (bht)wr[i].x;
            Bl[buf][(n4 + 1) * 40 + k] = (bht)wr[i].y;
            Bl[buf][(n4 + 2) * 40 + k] = (bht)wr[i].z;
            Bl[buf][(n4 + 3) * 40 + k] = (bht)wr[i].w;
        }
    };

    v8f acc[2][4] = {};
    const int nsteps = K / 32;

    // prologue: fill buffer 0
    issueA(0, 0);
    loadW(0);
    storeW(0);

    for (int i = 0; i < nsteps; ++i) {
        const int cur = i & 1;
        if (tid < 32) __builtin_amdgcn_s_wait_tensorcnt(0);
        __syncthreads();   // Al[cur]/Bl[cur] ready; prior reads of Al/Bl[cur^1] done

        // kick off next tile: DMA + W global loads overlap the WMMA block below
        if (i + 1 < nsteps) {
            issueA((i + 1) * 32, cur ^ 1);
            loadW((i + 1) * 32);
        }

        // ---- B fragments (shared by both M-subtiles) ----
        BFrag b[4];
        #pragma unroll
        for (int t = 0; t < 4; ++t) {
            const bht* p = &Bl[cur][(t * 16 + lm) * 40 + half * 16];
            b[t].q[0] = *(const uint4*)p;
            b[t].q[1] = *(const uint4*)(p + 8);
        }
        // ---- two 16-row A fragments, 8 WMMAs ----
        #pragma unroll
        for (int sub = 0; sub < 2; ++sub) {
            AFrag a;
            const int mrow = wave * 32 + sub * 16 + lm;
            #pragma unroll
            for (int vv = 0; vv < 8; ++vv) {
                int kp = half * 16 + (vv << 1);
                a.u[vv] = *(const unsigned*)&Al[cur][mrow * 40 + kp];
            }
            #pragma unroll
            for (int t = 0; t < 4; ++t)
                acc[sub][t] = WMMA_BF16(a.v, b[t].v, acc[sub][t]);
        }

        // convert + store next W tile (loadcnt wait lands after the WMMAs)
        if (i + 1 < nsteps) storeW(cur ^ 1);
    }

    // epilogue (C layout: vgpr r -> row r + 8*half, col = lane&15)
    #pragma unroll
    for (int sub = 0; sub < 2; ++sub) {
        #pragma unroll
        for (int t = 0; t < 4; ++t) {
            #pragma unroll
            for (int r = 0; r < 8; ++r) {
                int row = m0 + wave * 32 + sub * 16 + r + half * 8;
                int col = n0 + t * 16 + lm;
                size_t o = (size_t)row * N + col;
                float av = acc[sub][t][r];
                if constexpr (MODE == M_BF16) {
                    outB[o] = (bht)av;
                } else if constexpr (MODE == M_F32) {
                    outF[o] = av;
                } else if constexpr (MODE == M_RES) {
                    outF[o] = res[o] + av;
                } else if constexpr (MODE == M_RES_BIAS) {
                    outF[o] = res[o] + av + bias[col];
                } else { // M_GELU (exact)
                    float xg = av + bias[col];
                    outB[o] = (bht)(0.5f * xg * (1.0f + erff(xg * 0.70710678118654752f)));
                }
            }
        }
    }
}

// ---------------- flash attention (causal) ----------------
// q,k,v,y: bf16 [B, L, H*DH] row-major. block = (b, h, 64-row q tile), 4 waves.
__global__ __launch_bounds__(128)
void gpt_attn(const bht* __restrict__ q, const bht* __restrict__ k,
              const bht* __restrict__ v, bht* __restrict__ y)
{
    __shared__ bht Kl[32 * 72];       // [key][dh]   (dh-contiguous == B-frag layout)
    __shared__ bht Vl[64 * 40];       // [dh][key]   (transposed for P@V)
    __shared__ bht Pl[4][16 * 40];    // per-wave P tile

    const int tid  = threadIdx.x;
    const int wave = tid >> 5, lane = tid & 31, half = lane >> 4, lm = lane & 15;
    const int qt = blockIdx.x & 15;           // 16 q-tiles of 64
    const int bh = blockIdx.x >> 4;
    const int b  = bh >> 4, h = bh & 15;

    const size_t base = (size_t)b * VL * VD + (size_t)h * VDH;
    const bht* Qb = q + base;
    const bht* Kb = k + base;
    const bht* Vb = v + base;
    bht*       Yb = y + base;

    const int qrow0 = qt * 64 + wave * 16;

    // Q fragments for dh = 0..31 and 32..63 (loaded once, straight from global)
    AFrag qa[2];
    #pragma unroll
    for (int s = 0; s < 2; ++s)
        #pragma unroll
        for (int vv = 0; vv < 8; ++vv) {
            int kp = s * 32 + half * 16 + (vv << 1);
            qa[s].u[vv] = *(const unsigned*)(Qb + (size_t)(qrow0 + lm) * VD + kp);
        }

    v8f o[4] = {};
    float msl[8], lsl[8];
    #pragma unroll
    for (int r = 0; r < 8; ++r) { msl[r] = -1e30f; lsl[r] = 0.f; }

    const int nkt = qt * 2 + 2;               // key tiles of 32 up to causal limit
    for (int kt = 0; kt < nkt; ++kt) {
        const int kbase = kt * 32;
        __syncthreads();
        // stage K tile 32x64 (direct copy)
        #pragma unroll
        for (int i = 0; i < 2; ++i) {
            int c = i * 128 + tid;
            int key = c >> 3, off = (c & 7) * 8;
            *(uint4*)&Kl[key * 72 + off] =
                *(const uint4*)(Kb + (size_t)(kbase + key) * VD + off);
        }
        // stage V tile transposed: Vl[dh][key]
        {
            int key = tid >> 2, dh0 = (tid & 3) * 16;
            const bht* src = Vb + (size_t)(kbase + key) * VD + dh0;
            #pragma unroll
            for (int j = 0; j < 16; ++j) Vl[(dh0 + j) * 40 + key] = src[j];
        }
        __syncthreads();

        // S = Q @ K^T  (16q x 32keys per wave: 2 n-tiles x 2 dh-steps)
        v8f s0 = {}, s1 = {};
        #pragma unroll
        for (int s = 0; s < 2; ++s) {
            BFrag b0, b1;
            const bht* p0 = &Kl[lm * 72 + s * 32 + half * 16];
            const bht* p1 = &Kl[(16 + lm) * 72 + s * 32 + half * 16];
            b0.q[0] = *(const uint4*)p0; b0.q[1] = *(const uint4*)(p0 + 8);
            b1.q[0] = *(const uint4*)p1; b1.q[1] = *(const uint4*)(p1 + 8);
            s0 = WMMA_BF16(qa[s].v, b0.v, s0);
            s1 = WMMA_BF16(qa[s].v, b1.v, s1);
        }

        // online softmax: row = vgpr slot (+8*half), col = lane&15
        #pragma unroll
        for (int r = 0; r < 8; ++r) {
            const int qr = qrow0 + r + half * 8;
            float e0 = (kbase + lm      <= qr) ? s0[r] * 0.125f : -1e30f;
            float e1 = (kbase + 16 + lm <= qr) ? s1[r] * 0.125f : -1e30f;
            float mx = fmaxf(e0, e1);
            #pragma unroll
            for (int oo = 8; oo > 0; oo >>= 1) mx = fmaxf(mx, __shfl_xor(mx, oo, 32));
            float mn = fmaxf(msl[r], mx);
            float sc = __expf(msl[r] - mn);
            msl[r] = mn;
            float p0 = __expf(e0 - mn), p1 = __expf(e1 - mn);
            float rs = p0 + p1;
            #pragma unroll
            for (int oo = 8; oo > 0; oo >>= 1) rs += __shfl_xor(rs, oo, 32);
            lsl[r] = lsl[r] * sc + rs;
            o[0][r] *= sc; o[1][r] *= sc; o[2][r] *= sc; o[3][r] *= sc;
            int prow = r + half * 8;
            Pl[wave][prow * 40 + lm]      = (bht)p0;
            Pl[wave][prow * 40 + 16 + lm] = (bht)p1;
        }

        // P (16x32) as A-fragment via per-wave LDS round-trip
        AFrag pa;
        #pragma unroll
        for (int vv = 0; vv < 8; ++vv) {
            int kp = half * 16 + (vv << 1);
            pa.u[vv] = *(const unsigned*)&Pl[wave][lm * 40 + kp];
        }
        // O += P @ V  (4 dh n-tiles)
        #pragma unroll
        for (int t = 0; t < 4; ++t) {
            BFrag vb;
            const bht* p = &Vl[(t * 16 + lm) * 40 + half * 16];
            vb.q[0] = *(const uint4*)p; vb.q[1] = *(const uint4*)(p + 8);
            o[t] = WMMA_BF16(pa.v, vb.v, o[t]);
        }
    }

    #pragma unroll
    for (int r = 0; r < 8; ++r) {
        float inv = 1.0f / lsl[r];
        int row = qrow0 + r + half * 8;
        #pragma unroll
        for (int t = 0; t < 4; ++t)
            Yb[(size_t)row * VD + t * 16 + lm] = (bht)(o[t][r] * inv);
    }
}

// ---------------- host orchestration ----------------
extern "C" void kernel_launch(void* const* d_in, const int* in_sizes, int n_in,
                              void* d_out, int out_size, void* d_ws, size_t ws_size,
                              hipStream_t stream)
{
    const int*   idx  = (const int*)  d_in[0];
    const float* tok  = (const float*)d_in[1];
    const float* ln1w = (const float*)d_in[2];
    const float* ln1b = (const float*)d_in[3];
    const float* wq   = (const float*)d_in[4];
    const float* wk   = (const float*)d_in[5];
    const float* wv   = (const float*)d_in[6];
    const float* wo   = (const float*)d_in[7];
    const float* ln2w = (const float*)d_in[8];
    const float* ln2b = (const float*)d_in[9];
    const float* w1   = (const float*)d_in[10];
    const float* b1   = (const float*)d_in[11];
    const float* w2   = (const float*)d_in[12];
    const float* b2   = (const float*)d_in[13];
    const float* lnfw = (const float*)d_in[14];
    const float* lnfb = (const float*)d_in[15];
    const float* lmh  = (const float*)d_in[16];

    // workspace layout (bytes)
    const size_t SZ_X = (size_t)VM * VD * 4;   // fp32 residual stream
    const size_t SZ_H = (size_t)VM * VD * 2;   // bf16 LN output
    const size_t SZ_Q = (size_t)VM * VD * 2;
    const size_t SZ_U = (size_t)VM * VFF * 2;  // bf16 gelu output
    char* ws = (char*)d_ws;
    size_t off = 0;
    float* x  = (float*)(ws + off); off += SZ_X;
    bht* h    = (bht*)(ws + off);   off += SZ_H;
    bht* qb   = (bht*)(ws + off);   off += SZ_Q;
    bht* kb   = (bht*)(ws + off);   off += SZ_Q;
    bht* vb   = (bht*)(ws + off);   off += SZ_Q;
    bht* yb   = (bht*)(ws + off);   off += SZ_Q;
    bht* ub   = (bht*)(ws + off);   off += SZ_U;
    if (ws_size < off) return;     // insufficient scratch: do nothing (deterministic)

    const dim3 blkG(128), blkR(256);
    const dim3 gD (VD  / 64, VM / 128);   // N=1024 GEMMs
    const dim3 gFF(VFF / 64, VM / 128);   // N=4096 GEMM
    const dim3 gV (VVOC/ 64, VM / 128);   // N=32000 GEMM

    gpt_embed<<<VM, blkR, 0, stream>>>(idx, tok, x);

    for (int ll = 0; ll < VNL; ++ll) {
        const size_t oD  = (size_t)ll * VD;
        const size_t oDD = (size_t)ll * VD * VD;
        const size_t oDF = (size_t)ll * VD * VFF;

        gpt_ln<<<VM, blkR, 0, stream>>>(x, ln1w + oD, ln1b + oD, h);

        gpt_gemm<M_BF16><<<gD, blkG, 0, stream>>>(h, VD, wq + oDD, VD, nullptr, nullptr, nullptr, qb);
        gpt_gemm<M_BF16><<<gD, blkG, 0, stream>>>(h, VD, wk + oDD, VD, nullptr, nullptr, nullptr, kb);
        gpt_gemm<M_BF16><<<gD, blkG, 0, stream>>>(h, VD, wv + oDD, VD, nullptr, nullptr, nullptr, vb);

        gpt_attn<<<VB * VNH * (VL / 64), blkG, 0, stream>>>(qb, kb, vb, yb);

        gpt_gemm<M_RES><<<gD, blkG, 0, stream>>>(yb, VD, wo + oDD, VD, nullptr, x, x, nullptr);

        gpt_ln<<<VM, blkR, 0, stream>>>(x, ln2w + oD, ln2b + oD, h);

        gpt_gemm<M_GELU><<<gFF, blkG, 0, stream>>>(h, VD, w1 + oDF, VFF, b1 + (size_t)ll * VFF,
                                                   nullptr, nullptr, ub);
        gpt_gemm<M_RES_BIAS><<<gD, blkG, 0, stream>>>(ub, VFF, w2 + oDF, VD, b2 + oD, x, x, nullptr);
    }

    gpt_ln<<<VM, blkR, 0, stream>>>(x, lnfw, lnfb, h);
    gpt_gemm<M_F32><<<gV, blkG, 0, stream>>>(h, VD, lmh, VVOC, nullptr, nullptr,
                                             (float*)d_out, nullptr);
}
